// SelfAttention_4140348473869
// MI455X (gfx1250) — matrix operand order repaired and checked by reading, not compiled, not run on hardware
//
#include <hip/hip_runtime.h>
#include <stdint.h>

// ---------------------------------------------------------------------------
// Problem constants (match reference)
// ---------------------------------------------------------------------------
constexpr int Bc  = 8;       // batch
constexpr int Hh  = 8;       // heads
constexpr int SEQ = 1024;    // sequence length
constexpr int DH  = 256;     // head dim (== model dim D)
constexpr int FF  = Hh * DH; // 2048 concat-head feature dim

constexpr int P_STRIDE  = 40;  // P-tile LDS row stride (u16): 20-dword bank rotation
constexpr int K_STRIDE  = 264; // K-block LDS row stride (u16): 4-dword bank rotation
constexpr int KT_STRIDE = 40;  // KT-block LDS row stride (u16)

// ---------------------------------------------------------------------------
// Types & helpers
// ---------------------------------------------------------------------------
typedef __attribute__((ext_vector_type(16))) __bf16    v16bf;
typedef __attribute__((ext_vector_type(8)))  float     v8f;
typedef __attribute__((ext_vector_type(4)))  float     v4f;
typedef __attribute__((ext_vector_type(4)))  uint32_t  v4u;
typedef __attribute__((ext_vector_type(4)))  int       v4i;

__device__ inline v8f v8f_zero() {
    v8f r;
    #pragma unroll
    for (int i = 0; i < 8; ++i) r[i] = 0.0f;
    return r;
}

// native f32 -> bf16
__device__ inline uint16_t bf16_1(float f) {
    __bf16 b = (__bf16)f;
    return __builtin_bit_cast(uint16_t, b);
}
__device__ inline uint32_t cvt2_bf16(float lo, float hi) {
    return (uint32_t)bf16_1(lo) | ((uint32_t)bf16_1(hi) << 16);
}

__device__ inline float fast_exp2(float x) {
#if __has_builtin(__builtin_amdgcn_exp2f)
    return __builtin_amdgcn_exp2f(x);
#else
    return exp2f(x);
#endif
}

// Build a v16bf A/B operand from two 16-byte (8 x bf16) chunks in memory/LDS.
__device__ inline v16bf load16bf(const uint16_t* p0, const uint16_t* p1) {
    union { v4u q[2]; v16bf v; } u;
    u.q[0] = *(const v4u*)p0;
    u.q[1] = *(const v4u*)p1;
    return u.v;
}

__device__ inline v8f wmma_bf16(v16bf a, v16bf b, v8f c) {
    return __builtin_amdgcn_wmma_f32_16x16x32_bf16(false, a, false, b,
                                                   (short)0, c, false, false);
}

// Async global -> LDS copy of one 16-byte granule (ASYNCcnt-tracked on CDNA5).
// Builtin signature (from clang diagnostic): (AS1 int4*, AS3 int4*, imm, imm).
__device__ inline void async_copy_b128(const uint16_t* gsrc, uint16_t* ldst) {
#if __has_builtin(__builtin_amdgcn_global_load_async_to_lds_b128)
    __builtin_amdgcn_global_load_async_to_lds_b128(
        (__attribute__((address_space(1))) v4i*)gsrc,
        (__attribute__((address_space(3))) v4i*)ldst, 0, 0);
#else
    *(v4u*)ldst = *(const v4u*)gsrc;
#endif
}

template <int N>
__device__ inline void wait_async_n() {
#if __has_builtin(__builtin_amdgcn_s_wait_asynccnt)
    __builtin_amdgcn_s_wait_asynccnt(N);
#else
    asm volatile("s_wait_asynccnt %0" :: "i"(N) : "memory");
#endif
}

// ---------------------------------------------------------------------------
// Pre-pass 0a: elementwise f32 -> bf16 (x). 8 elems/thread.
// ---------------------------------------------------------------------------
__global__ void __launch_bounds__(256)
convert_f32_bf16(const float* __restrict__ src, uint16_t* __restrict__ dst) {
    const size_t i = ((size_t)blockIdx.x * 256 + threadIdx.x) * 8;
    #pragma unroll
    for (int j = 0; j < 8; j += 2)
        *(uint32_t*)&dst[i + j] = cvt2_bf16(src[i + j], src[i + j + 1]);
}

// ---------------------------------------------------------------------------
// Pre-pass 0b: transpose + convert.  out[g][c][r] = (bf16) in[g][r][c]
// ---------------------------------------------------------------------------
__global__ void __launch_bounds__(256)
transpose_convert(const float* __restrict__ src, uint16_t* __restrict__ dst,
                  int R, int C) {
    const size_t idx = (size_t)blockIdx.x * 256 + threadIdx.x;
    const int r = (int)(idx % R);
    const size_t t = idx / R;
    const int c = (int)(t % C);
    const int g = (int)(t / C);
    dst[idx] = bf16_1(src[((size_t)g * R + r) * C + c]);
}

// ---------------------------------------------------------------------------
// Kernel 1: Q/K projection, no LDS (asm verified optimal in round 2).
// ---------------------------------------------------------------------------
__global__ void __launch_bounds__(256)
proj_qk_kernel(const uint16_t* __restrict__ xb, const uint16_t* __restrict__ wq_t,
               const uint16_t* __restrict__ wk_t, uint16_t* __restrict__ Qb,
               uint16_t* __restrict__ Kb, uint16_t* __restrict__ KTb) {
    const int lane = threadIdx.x & 31;
    const int wave = threadIdx.x >> 5;
    const int hi   = lane >> 4;
    const int ln   = lane & 15;

    const int nBase = blockIdx.x * 128;
    const int mBase = blockIdx.y * 64;
    const int g  = blockIdx.z & 1;          // 0 -> Q, 1 -> K
    const int bh = blockIdx.z >> 1;
    const int b  = bh >> 3;
    const int h  = bh & 7;

    const uint16_t* Wh  = (g ? wk_t : wq_t) + (size_t)h * DH * DH; // [e][k]
    uint16_t*       Out = (g ? Kb : Qb) + (size_t)bh * SEQ * DH;

    const int mSub = (wave & 3) * 16;
    const int nSub = (wave >> 2) * 64;

    const uint16_t* Arow = xb + ((size_t)b * SEQ + mBase + mSub + ln) * DH;
    const uint16_t* Brow[4];
    #pragma unroll
    for (int nt = 0; nt < 4; ++nt)
        Brow[nt] = Wh + (size_t)(nBase + nSub + nt * 16 + ln) * DH;

    const int koffA = hi * 16;
    const int koffB = hi * 16;

    v8f acc[4];
    #pragma unroll
    for (int i = 0; i < 4; ++i) acc[i] = v8f_zero();

    for (int kk = 0; kk < DH; kk += 32) {
        v16bf a  = load16bf(Arow + kk + koffA, Arow + kk + 8 + koffA);
        v16bf b0 = load16bf(Brow[0] + kk + koffB, Brow[0] + kk + koffB + 8);
        v16bf b1 = load16bf(Brow[1] + kk + koffB, Brow[1] + kk + koffB + 8);
        v16bf b2 = load16bf(Brow[2] + kk + koffB, Brow[2] + kk + koffB + 8);
        v16bf b3 = load16bf(Brow[3] + kk + koffB, Brow[3] + kk + koffB + 8);
        acc[0] = wmma_bf16(a, b0, acc[0]);
        acc[1] = wmma_bf16(a, b1, acc[1]);
        acc[2] = wmma_bf16(a, b2, acc[2]);
        acc[3] = wmma_bf16(a, b3, acc[3]);
    }

    #pragma unroll
    for (int nt = 0; nt < 4; ++nt) {
        const int e = nBase + nSub + nt * 16 + ln;
        #pragma unroll
        for (int r = 0; r < 8; ++r) {
            const int srow = mBase + mSub + r + 8 * hi;
            Out[(size_t)srow * DH + e] = bf16_1(acc[nt][r]);
        }
    }
    if (g == 1) {
        const int s0 = mBase + mSub + 8 * hi;
        #pragma unroll
        for (int nt = 0; nt < 4; ++nt) {
            const int e = nBase + nSub + nt * 16 + ln;
            v4u pk;
            #pragma unroll
            for (int j = 0; j < 4; ++j)
                pk[j] = cvt2_bf16(acc[nt][2 * j], acc[nt][2 * j + 1]);
            *(v4u*)&KTb[((size_t)bh * DH + e) * SEQ + s0] = pk;
        }
    }
}

// ---------------------------------------------------------------------------
// Kernel 2: flash attention per (b,h). V == K (reference bug).
//   Double-buffered async global->LDS staging of K / KT blocks (ASYNCcnt),
//   shared by all 8 waves; Q fragments held in registers throughout.
// ---------------------------------------------------------------------------
__global__ void __launch_bounds__(256)
attn_kernel(const uint16_t* __restrict__ Qb, const uint16_t* __restrict__ Kb,
            const uint16_t* __restrict__ KTb, uint16_t* __restrict__ Ob) {
    const int lane = threadIdx.x & 31;
    const int wave = threadIdx.x >> 5;
    const int hi   = lane >> 4;
    const int ln   = lane & 15;
    const int tid  = threadIdx.x;

    const int bh = blockIdx.y;
    const int b  = bh >> 3;
    const int h  = bh & 7;
    const int q0 = blockIdx.x * 128 + wave * 16;

    const uint16_t* Qh  = Qb  + (size_t)bh * SEQ * DH;
    const uint16_t* Kh  = Kb  + (size_t)bh * SEQ * DH;
    const uint16_t* KTh = KTb + (size_t)bh * DH * SEQ;

    __shared__ __align__(16) uint16_t Kblk[2][32 * K_STRIDE];    // [key][e]
    __shared__ __align__(16) uint16_t KTblk[2][DH * KT_STRIDE];  // [e][key]
    __shared__ __align__(16) uint16_t Pt[8 * 16 * P_STRIDE];
    uint16_t* Pw = &Pt[wave * 16 * P_STRIDE];

    // ---- stage one 32-key block (8 async b128 granules per thread) ----
    auto stage = [&](int k0, int buf) {
        const int key = tid >> 3;               // 0..31
        const int c   = (tid & 7) * 32;         // 0..224
        const uint16_t* ksrc = Kh + (size_t)(k0 + key) * DH + c;
        uint16_t*       kdst = &Kblk[buf][key * K_STRIDE + c];
        async_copy_b128(ksrc,      kdst);
        async_copy_b128(ksrc + 8,  kdst + 8);
        async_copy_b128(ksrc + 16, kdst + 16);
        async_copy_b128(ksrc + 24, kdst + 24);
        const uint16_t* tsrc = KTh + (size_t)tid * SEQ + k0;     // e = tid
        uint16_t*       tdst = &KTblk[buf][tid * KT_STRIDE];
        async_copy_b128(tsrc,      tdst);
        async_copy_b128(tsrc + 8,  tdst + 8);
        async_copy_b128(tsrc + 16, tdst + 16);
        async_copy_b128(tsrc + 24, tdst + 24);
    };

    // ---- preload the wave's Q tile (16 x 256) into registers ----
    const uint16_t* Arow = Qh + (size_t)(q0 + ln) * DH;
    const int koffA = hi * 16;
    v16bf aQ[8];
    #pragma unroll
    for (int e8 = 0; e8 < 8; ++e8)
        aQ[e8] = load16bf(Arow + e8 * 32 + koffA, Arow + e8 * 32 + 8 + koffA);

    v8f o[16];
    #pragma unroll
    for (int i = 0; i < 16; ++i) o[i] = v8f_zero();
    float mrow[8], lrow[8];
    #pragma unroll
    for (int r = 0; r < 8; ++r) { mrow[r] = -3.0e38f; lrow[r] = 0.0f; }

    // softmax in exp2 domain: scale = 1/sqrt(256) * log2(e)
    const float sc2 = 0.0625f * 1.44269504088896f;

    stage(0, 0);

    for (int k0 = 0; k0 < SEQ; k0 += 32) {
        const int cur = (k0 >> 5) & 1;
        const int nxt = cur ^ 1;
        const bool more = (k0 + 32) < SEQ;

        __syncthreads();                 // everyone done reading buf[nxt]
        if (more) stage(k0 + 32, nxt);
        if (more) wait_async_n<8>();     // own cur-block granules landed
        else      wait_async_n<0>();
        __syncthreads();                 // all waves' cur data visible

        // --- scores: 16 queries x 32 keys from LDS K block ---
        v8f s0 = v8f_zero(), s1 = v8f_zero();
        const uint16_t* kr0 = &Kblk[cur][ln * K_STRIDE + hi * 16];
        const uint16_t* kr1 = &Kblk[cur][(16 + ln) * K_STRIDE + hi * 16];
        #pragma unroll
        for (int e8 = 0; e8 < 8; ++e8) {
            v16bf b0 = load16bf(kr0 + e8 * 32, kr0 + e8 * 32 + 8);
            v16bf b1 = load16bf(kr1 + e8 * 32, kr1 + e8 * 32 + 8);
            s0 = wmma_bf16(aQ[e8], b0, s0);
            s1 = wmma_bf16(aQ[e8], b1, s1);
        }

        // --- online softmax (row = r + 8*hi lives in lanes hi*16..hi*16+15) ---
        #pragma unroll
        for (int r = 0; r < 8; ++r) {
            float v0 = s0[r] * sc2;
            float v1 = s1[r] * sc2;
            float mx = fmaxf(v0, v1);
            #pragma unroll
            for (int off = 1; off < 16; off <<= 1)
                mx = fmaxf(mx, __shfl_xor(mx, off, 32));
            float mnew = fmaxf(mrow[r], mx);
            float corr = fast_exp2(mrow[r] - mnew);
            float p0 = fast_exp2(v0 - mnew);
            float p1 = fast_exp2(v1 - mnew);
            float sum = p0 + p1;
            #pragma unroll
            for (int off = 1; off < 16; off <<= 1)
                sum += __shfl_xor(sum, off, 32);
            lrow[r] = lrow[r] * corr + sum;
            mrow[r] = mnew;
            #pragma unroll
            for (int nt = 0; nt < 16; ++nt) o[nt][r] *= corr;
            const int prow = r + 8 * hi;
            Pw[prow * P_STRIDE + ln]      = bf16_1(p0);
            Pw[prow * P_STRIDE + 16 + ln] = bf16_1(p1);
        }
        asm volatile("s_wait_dscnt 0" ::: "memory");

        // reload P in A-operand layout (per-wave tile, same-wave DS in-order)
        const uint16_t* pr = &Pw[ln * P_STRIDE + koffA];
        v16bf aP = load16bf(pr, pr + 8);

        // --- O += P * V (V = K): B-operand rows from LDS KT block ---
        #pragma unroll
        for (int ntg = 0; ntg < 4; ++ntg) {
            v16bf bv[4];
            #pragma unroll
            for (int j = 0; j < 4; ++j) {
                const uint16_t* br =
                    &KTblk[cur][((ntg * 4 + j) * 16 + ln) * KT_STRIDE + hi * 16];
                bv[j] = load16bf(br, br + 8);
            }
            #pragma unroll
            for (int j = 0; j < 4; ++j)
                o[ntg * 4 + j] = wmma_bf16(aP, bv[j], o[ntg * 4 + j]);
        }
    }

    // --- normalize and store O as [B, S, H*D] bf16 ---
    #pragma unroll
    for (int r = 0; r < 8; ++r) {
        const float inv = 1.0f / lrow[r];
        const int q = q0 + r + 8 * hi;
        const size_t base = (size_t)(b * SEQ + q) * FF + (size_t)h * DH;
        #pragma unroll
        for (int nt = 0; nt < 16; ++nt)
            Ob[base + nt * 16 + ln] = bf16_1(o[nt][r] * inv);
    }
}

// ---------------------------------------------------------------------------
// Kernel 3: output projection, transposed:  out^T[d][q] = wo_t[d][:] . Ob[q][:]
// ---------------------------------------------------------------------------
__global__ void __launch_bounds__(256)
out_projT_kernel(const uint16_t* __restrict__ Ob, const uint16_t* __restrict__ wo_t,
                 float* __restrict__ out) {
    const int lane = threadIdx.x & 31;
    const int wave = threadIdx.x >> 5;
    const int hi   = lane >> 4;
    const int ln   = lane & 15;

    const int nBase = blockIdx.x * 128;   // q
    const int mBase = blockIdx.y * 64;    // d

    const int mSub = (wave & 3) * 16;
    const int nSub = (wave >> 2) * 64;

    const uint16_t* Arow = wo_t + (size_t)(mBase + mSub + ln) * FF;
    const uint16_t* Brow[4];
    #pragma unroll
    for (int nt = 0; nt < 4; ++nt)
        Brow[nt] = Ob + (size_t)(nBase + nSub + nt * 16 + ln) * FF;

    const int koffA = hi * 16;
    const int koffB = hi * 16;

    v8f acc[4];
    #pragma unroll
    for (int i = 0; i < 4; ++i) acc[i] = v8f_zero();

    for (int kk = 0; kk < FF; kk += 32) {
        v16bf a  = load16bf(Arow + kk + koffA, Arow + kk + 8 + koffA);
        v16bf b0 = load16bf(Brow[0] + kk + koffB, Brow[0] + kk + koffB + 8);
        v16bf b1 = load16bf(Brow[1] + kk + koffB, Brow[1] + kk + koffB + 8);
        v16bf b2 = load16bf(Brow[2] + kk + koffB, Brow[2] + kk + koffB + 8);
        v16bf b3 = load16bf(Brow[3] + kk + koffB, Brow[3] + kk + koffB + 8);
        acc[0] = wmma_bf16(a, b0, acc[0]);
        acc[1] = wmma_bf16(a, b1, acc[1]);
        acc[2] = wmma_bf16(a, b2, acc[2]);
        acc[3] = wmma_bf16(a, b3, acc[3]);
    }

    const int d0 = mBase + mSub + 8 * hi;
    #pragma unroll
    for (int nt = 0; nt < 4; ++nt) {
        const int q = nBase + nSub + nt * 16 + ln;
        v4f lo, hi4;
        #pragma unroll
        for (int j = 0; j < 4; ++j) { lo[j] = acc[nt][j]; hi4[j] = acc[nt][4 + j]; }
        *(v4f*)(out + (size_t)q * DH + d0)     = lo;
        *(v4f*)(out + (size_t)q * DH + d0 + 4) = hi4;
    }
}

// ---------------------------------------------------------------------------
// Launch. Workspace (~135 MB) fits the 192 MB L2.
// ---------------------------------------------------------------------------
extern "C" void kernel_launch(void* const* d_in, const int* in_sizes, int n_in,
                              void* d_out, int out_size, void* d_ws, size_t ws_size,
                              hipStream_t stream) {
    (void)in_sizes; (void)n_in; (void)out_size; (void)ws_size;

    const float* x  = (const float*)d_in[0];
    const float* wq = (const float*)d_in[1];
    const float* wk = (const float*)d_in[2];
    // d_in[3] (wv) is dead per the reference bug: V is projected with wk.
    const float* wo = (const float*)d_in[4];
    float* out = (float*)d_out;

    const size_t XN = (size_t)Bc * SEQ * DH;
    const size_t WN = (size_t)Hh * DH * DH;
    const size_t QN = (size_t)Bc * Hh * SEQ * DH;

    uint16_t* xb   = (uint16_t*)d_ws;
    uint16_t* wq_t = xb + XN;
    uint16_t* wk_t = wq_t + WN;
    uint16_t* wo_t = wk_t + WN;                     // [DH][FF]
    uint16_t* Qb   = wo_t + WN;
    uint16_t* Kb   = Qb + QN;
    uint16_t* KTb  = Kb + QN;
    uint16_t* Ob   = KTb + QN;

    convert_f32_bf16<<<dim3((unsigned)(XN / 2048)), 256, 0, stream>>>(x, xb);
    transpose_convert<<<dim3((unsigned)(WN / 256)), 256, 0, stream>>>(wq, wq_t, DH, DH);
    transpose_convert<<<dim3((unsigned)(WN / 256)), 256, 0, stream>>>(wk, wk_t, DH, DH);
    transpose_convert<<<dim3((unsigned)(WN / 256)), 256, 0, stream>>>(wo, wo_t, FF, DH);

    proj_qk_kernel<<<dim3(DH / 128, SEQ / 64, Bc * Hh * 2), 256, 0, stream>>>(
        xb, wq_t, wk_t, Qb, Kb, KTb);
    attn_kernel<<<dim3(SEQ / 128, Bc * Hh), 256, 0, stream>>>(Qb, Kb, KTb, Ob);
    out_projT_kernel<<<dim3((Bc * SEQ) / 128, DH / 64), 256, 0, stream>>>(
        Ob, wo_t, out);
}
